// Grok1MoE_18210661335575
// MI455X (gfx1250) — matrix operand order repaired and checked by reading, not compiled, not run on hardware
//
#include <hip/hip_runtime.h>
#include <hip/hip_bf16.h>
#include <math.h>

// Grok1 MoE: T=4096 tokens, H=1024 hidden, I=4096 intermediate, E=8 experts, top-2.
// Sparse (gathered) expert compute with v_wmma_f32_16x16x32_bf16 matrix cores.
// Weights pre-converted to bf16 in ws (once per launch) when ws is big enough;
// otherwise converted in-loop via v_cvt_pk_bf16_f32.

#define TOK   4096
#define HID   1024
#define INTER 4096
#define NEXP  8
#define SCAP  30.0f

typedef __attribute__((ext_vector_type(16))) __bf16 v16bf;
typedef __attribute__((ext_vector_type(8)))  float  v8f;

union BF16Frag {
    v16bf v;
    __bf16 h[16];
    unsigned short u[16];
};

__device__ __forceinline__ unsigned short bf_bits(float x) {
    __bf16 b = (__bf16)x;                       // RNE, native cvt on gfx1250
    return __builtin_bit_cast(unsigned short, b);
}

__device__ __forceinline__ v8f wmma_bf16(const BF16Frag& a, const BF16Frag& b, v8f c) {
    // D = A(16x32 bf16) * B(32x16 bf16) + C(16x16 f32)
    return __builtin_amdgcn_wmma_f32_16x16x32_bf16(
        /*neg_a=*/false, a.v, /*neg_b=*/false, b.v,
        /*c_mod=*/(short)0, c, /*reuse_a=*/false, /*reuse_b=*/false);
}

// B-fragment loader: WT=float -> hardware packed convert; WT=__bf16 -> bit copy.
template <typename WT>
__device__ __forceinline__ void load_bfrag(BF16Frag& b, const WT* __restrict__ p) {
#pragma unroll
    for (int j = 0; j < 16; ++j) b.h[j] = (__bf16)p[j];
}

// ---------------------------------------------------------------------------
// Kernel 0: zero output (harness poisons it) + per-expert counters
// ---------------------------------------------------------------------------
__global__ void zero_out_kernel(float* __restrict__ out, int n, int* __restrict__ cnt) {
    int i = blockIdx.x * blockDim.x + threadIdx.x;
    if (i < n) out[i] = 0.0f;
    if (i < NEXP) cnt[i] = 0;
}

// ---------------------------------------------------------------------------
// Kernel 0b: bulk f32 -> bf16 weight conversion (bandwidth-bound, 8 elem/thread)
// ---------------------------------------------------------------------------
__global__ __launch_bounds__(256) void cvt_bf16_kernel(
    const float* __restrict__ src, __bf16* __restrict__ dst, int n)
{
    int i = (blockIdx.x * blockDim.x + threadIdx.x) * 8;
    if (i + 8 > n) return;                 // n is a multiple of 8 here anyway
    float4 a = *reinterpret_cast<const float4*>(src + i);
    float4 c = *reinterpret_cast<const float4*>(src + i + 4);
    union { __bf16 h[8]; uint4 u; } o;
    o.h[0] = (__bf16)a.x; o.h[1] = (__bf16)a.y; o.h[2] = (__bf16)a.z; o.h[3] = (__bf16)a.w;
    o.h[4] = (__bf16)c.x; o.h[5] = (__bf16)c.y; o.h[6] = (__bf16)c.z; o.h[7] = (__bf16)c.w;
    *reinterpret_cast<uint4*>(dst + i) = o.u;
}

// ---------------------------------------------------------------------------
// Kernel 1: router — fp32 logits + tanh softcap + softmax + top-2 scatter
// ---------------------------------------------------------------------------
__global__ __launch_bounds__(256) void router_kernel(
    const float* __restrict__ hidden, const float* __restrict__ gate_w,
    int* __restrict__ cnt, int* __restrict__ tok_list, float* __restrict__ wt_list)
{
    int t = blockIdx.x * blockDim.x + threadIdx.x;
    if (t >= TOK) return;
    const float* hrow = hidden + (size_t)t * HID;

    float acc[NEXP];
#pragma unroll
    for (int e = 0; e < NEXP; ++e) acc[e] = 0.0f;
    for (int h = 0; h < HID; ++h) {
        float x = hrow[h];
#pragma unroll
        for (int e = 0; e < NEXP; ++e) acc[e] += x * gate_w[e * HID + h];
    }
    float mx = -1e30f;
#pragma unroll
    for (int e = 0; e < NEXP; ++e) {
        acc[e] = SCAP * tanhf(acc[e] * (1.0f / SCAP));
        mx = fmaxf(mx, acc[e]);
    }
    float sum = 0.0f;
#pragma unroll
    for (int e = 0; e < NEXP; ++e) { acc[e] = expf(acc[e] - mx); sum += acc[e]; }
    float inv = 1.0f / sum;
#pragma unroll
    for (int e = 0; e < NEXP; ++e) acc[e] *= inv;

    int i1 = 0;
#pragma unroll
    for (int e = 1; e < NEXP; ++e) if (acc[e] > acc[i1]) i1 = e;
    int i2 = (i1 == 0) ? 1 : 0;
#pragma unroll
    for (int e = 0; e < NEXP; ++e) if (e != i1 && acc[e] > acc[i2]) i2 = e;

    int s1 = atomicAdd(&cnt[i1], 1);
    tok_list[i1 * TOK + s1] = t;  wt_list[i1 * TOK + s1] = acc[i1];
    int s2 = atomicAdd(&cnt[i2], 1);
    tok_list[i2 * TOK + s2] = t;  wt_list[i2 * TOK + s2] = acc[i2];
}

// ---------------------------------------------------------------------------
// Kernel 2: fused expert — (gelu(x w1^T) * (x w3^T)) w2^T, gathered tokens
//   block = 8 waves, one 16-token tile of one expert
//   Preload: gathered hidden tile (16 x H) -> LDS bf16 (convert once)
//   Phase A: act tile (16 x 1024 I-chunk) via WMMA -> LDS bf16
//   Phase B: down-proj partial (K = I-chunk) into persistent f32 accumulators
// ---------------------------------------------------------------------------
constexpr int ICHUNK = 1024;
constexpr int LDSS   = ICHUNK + 8;   // row stride (elems): 2064B -> /4 = 516 ≡ 4 mod 64

template <typename WT>
__global__ __launch_bounds__(256) void expert_kernel(
    const float* __restrict__ hidden,
    const WT* __restrict__ w1, const WT* __restrict__ w3,
    const WT* __restrict__ w2,
    const int* __restrict__ cnt, const int* __restrict__ tok_list,
    const float* __restrict__ wt_list, float* __restrict__ out)
{
    __shared__ unsigned short hid_lds[16 * LDSS];  // ~33 KB, gathered token tile
    __shared__ unsigned short act_lds[16 * LDSS];  // ~33 KB, gelu*up tile

    const int e  = blockIdx.y;
    const int tt = blockIdx.x;
    const int ne = cnt[e];
    if (tt * 16 >= ne) return;   // uniform early-exit

    const int tid  = threadIdx.x;
    const int wave = tid >> 5;       // 0..7
    const int lane = tid & 31;
    const int half = lane >> 4;      // lane half: selects K sub-range / M offset
    const int m    = lane & 15;      // A row / B column / C column

    // -------- Preload: gather + convert hidden tile into LDS (once) --------
    for (int idx = tid; idx < 16 * HID; idx += 256) {
        const int row  = idx >> 10;        // 0..15  (HID == 1024)
        const int col  = idx & (HID - 1);
        const int slot = tt * 16 + row;
        float v = 0.0f;
        if (slot < ne)
            v = hidden[(size_t)tok_list[e * TOK + slot] * HID + col];
        hid_lds[row * LDSS + col] = bf_bits(v);
    }
    __syncthreads();

    v8f oacc[8];
#pragma unroll
    for (int g = 0; g < 8; ++g)
#pragma unroll
        for (int r = 0; r < 8; ++r) oacc[g][r] = 0.0f;

    for (int ic = 0; ic < INTER / ICHUNK; ++ic) {
        // -------- Phase A: gate/up GEMM + gelu*mul -> LDS --------
#pragma unroll
        for (int grp = 0; grp < 2; ++grp) {
            const int i0 = ic * ICHUNK + wave * 128 + grp * 64;  // 4 N-tiles
            v8f ag[4], au[4];
#pragma unroll
            for (int nt = 0; nt < 4; ++nt)
#pragma unroll
                for (int r = 0; r < 8; ++r) { ag[nt][r] = 0.0f; au[nt][r] = 0.0f; }

            for (int kk = 0; kk < HID; kk += 32) {
                // A fragment: pure bit-copies from LDS (no conversion)
                BF16Frag a;
                const unsigned short* lrow = &hid_lds[m * LDSS];
                const int k0 = kk + half * 16;       // V0..V3: K = k0..k0+7
                const int k1 = kk + half * 16 + 8;   // V4..V7: K = k1..k1+7
#pragma unroll
                for (int j = 0; j < 8; ++j) { a.u[j] = lrow[k0 + j]; a.u[8 + j] = lrow[k1 + j]; }
#pragma unroll
                for (int nt = 0; nt < 4; ++nt) {
                    // B[k][n] = w{1,3}[e][i0+nt*16+n][kk+k]; lane reads 16 consecutive elems
                    const size_t woff =
                        ((size_t)e * INTER + (size_t)(i0 + nt * 16 + m)) * HID
                        + (size_t)(kk + half * 16);
                    BF16Frag b;
                    load_bfrag(b, w1 + woff);
                    ag[nt] = wmma_bf16(a, b, ag[nt]);
                    load_bfrag(b, w3 + woff);
                    au[nt] = wmma_bf16(a, b, au[nt]);
                }
            }
            // act = gelu_exact(gate) * up -> LDS (C layout: M=r+8*half, N=m)
#pragma unroll
            for (int nt = 0; nt < 4; ++nt)
#pragma unroll
                for (int r = 0; r < 8; ++r) {
                    float gv   = ag[nt][r];
                    float actv = 0.5f * gv * (1.0f + erff(gv * 0.70710678118f)) * au[nt][r];
                    int row = r + half * 8;
                    int col = wave * 128 + grp * 64 + nt * 16 + m;
                    act_lds[row * LDSS + col] = bf_bits(actv);
                }
        }
        __syncthreads();

        // -------- Phase B: down-proj partial over this I-chunk --------
        const int h0 = wave * 128;   // this wave owns 128 H columns
        for (int kk = 0; kk < ICHUNK; kk += 32) {
            BF16Frag a;
            const unsigned short* lrow = &act_lds[m * LDSS];
            const int k0 = kk + half * 16;
            const int k1 = kk + half * 16 + 8;
#pragma unroll
            for (int j = 0; j < 8; ++j) { a.u[j] = lrow[k0 + j]; a.u[8 + j] = lrow[k1 + j]; }
#pragma unroll
            for (int nt = 0; nt < 8; ++nt) {
                // B[k][n] = w2[e][h0+nt*16+n][ic*ICHUNK+kk+k]
                const size_t woff =
                    ((size_t)e * HID + (size_t)(h0 + nt * 16 + m)) * INTER
                    + (size_t)(ic * ICHUNK + kk + half * 16);
                BF16Frag b;
                load_bfrag(b, w2 + woff);
                oacc[nt] = wmma_bf16(a, b, oacc[nt]);
            }
        }
        __syncthreads();
    }

    // -------- Combine: scale by router prob, atomic-accumulate into out --------
#pragma unroll
    for (int r = 0; r < 8; ++r) {
        const int row  = r + half * 8;
        const int slot = tt * 16 + row;
        if (slot < ne) {
            const int   t = tok_list[e * TOK + slot];
            const float w = wt_list[e * TOK + slot];
#pragma unroll
            for (int nt = 0; nt < 8; ++nt) {
                const int h = wave * 128 + nt * 16 + m;
                atomicAdd(out + (size_t)t * HID + h, oacc[nt][r] * w);
            }
        }
    }
}

// ---------------------------------------------------------------------------
extern "C" void kernel_launch(void* const* d_in, const int* in_sizes, int n_in,
                              void* d_out, int out_size, void* d_ws, size_t ws_size,
                              hipStream_t stream) {
    (void)in_sizes; (void)n_in; (void)out_size;

    const float* hidden = (const float*)d_in[0];
    const float* gate_w = (const float*)d_in[1];
    const float* w1     = (const float*)d_in[2];
    const float* w3     = (const float*)d_in[3];
    const float* w2     = (const float*)d_in[4];
    float* out = (float*)d_out;

    // ws layout: cnt[8] @0; tok_list[E][T] i32 @1KiB; wt_list[E][T] f32;
    // optional bf16 weight mirrors @512KiB (3 x 64 MiB).
    char*  ws       = (char*)d_ws;
    int*   cnt      = (int*)ws;
    int*   tok_list = (int*)(ws + 1024);
    float* wt_list  = (float*)(ws + 1024 + sizeof(int) * NEXP * TOK);

    const size_t NW     = (size_t)NEXP * INTER * HID;       // elems per weight tensor
    const size_t WBYTES = NW * sizeof(unsigned short);      // 64 MiB
    const size_t WBASE  = 512 * 1024;
    const bool   pre    = ws_size >= WBASE + 3 * WBYTES;    // ~192.5 MiB needed

    const int n = TOK * HID;
    zero_out_kernel<<<(n + 255) / 256, 256, 0, stream>>>(out, n, cnt);
    router_kernel<<<TOK / 256, 256, 0, stream>>>(hidden, gate_w, cnt, tok_list, wt_list);

    dim3 grid(TOK / 16, NEXP);   // blocks beyond cnt[e] exit immediately
    if (pre) {
        __bf16* w1b = (__bf16*)(ws + WBASE);
        __bf16* w3b = (__bf16*)(ws + WBASE + WBYTES);
        __bf16* w2b = (__bf16*)(ws + WBASE + 2 * WBYTES);
        const int ncvt = (int)(NW / 8);                      // threads per tensor
        cvt_bf16_kernel<<<(ncvt + 255) / 256, 256, 0, stream>>>(w1, w1b, (int)NW);
        cvt_bf16_kernel<<<(ncvt + 255) / 256, 256, 0, stream>>>(w3, w3b, (int)NW);
        cvt_bf16_kernel<<<(ncvt + 255) / 256, 256, 0, stream>>>(w2, w2b, (int)NW);
        expert_kernel<__bf16><<<grid, 256, 0, stream>>>(hidden, w1b, w3b, w2b,
                                                        cnt, tok_list, wt_list, out);
    } else {
        expert_kernel<float><<<grid, 256, 0, stream>>>(hidden, w1, w3, w2,
                                                       cnt, tok_list, wt_list, out);
    }
}
